// RNNLayer_57724360458877
// MI455X (gfx1250) — matrix operand order repaired and checked by reading, not compiled, not run on hardware
//
#include <hip/hip_runtime.h>
#include <hip/hip_bf16.h>

#define D     1024
#define SEQ   2048
#define BATCH 32
#define ROWS  (SEQ * BATCH)   // 65536
#define LN_EPS 1e-5f

typedef __attribute__((ext_vector_type(16))) __bf16       v16bf;
typedef __attribute__((ext_vector_type(8)))  float        v8f;
typedef __attribute__((ext_vector_type(4)))  unsigned int v4u;

union V16 { v16bf v; v4u u[2]; };

__device__ __forceinline__ unsigned short f2bf(float f) {
    unsigned int u = __float_as_uint(f);
    u += 0x7fffu + ((u >> 16) & 1u);          // round-to-nearest-even
    return (unsigned short)(u >> 16);
}
__device__ __forceinline__ float bf2f(unsigned short h) {
    return __uint_as_float(((unsigned int)h) << 16);
}

__device__ __forceinline__ v8f wmma_bf16(V16 a, V16 b, v8f c) {
    // v_wmma_f32_16x16x32_bf16: D = A(16x32) * B(32x16) + C(16x16 f32)
    return __builtin_amdgcn_wmma_f32_16x16x32_bf16(
        /*neg_a=*/false, a.v, /*neg_b=*/false, b.v,
        /*c_mod=*/(short)0, c, /*reuse_a=*/false, /*reuse_b=*/false);
}

// A operand (16x32 bf16, M x K) from a row-major [16][D] bf16 tile in LDS.
// ISA layout: lanes 0-15 hold K[0..7],K[16..23]; lanes 16-31 hold K[8..15],K[24..31].
__device__ __forceinline__ V16 load_a_lds(const unsigned short* t, int k0, int lane) {
    const int row = lane & 15, kh = lane >> 4;
    const unsigned short* p = t + row * D + k0 + kh * 8;
    V16 a;
    a.u[0] = *(const v4u*)(p);        // K = k0 + kh*8 .. +7
    a.u[1] = *(const v4u*)(p + 16);   // K = k0 + 16 + kh*8 .. +7
    return a;
}

// B operand (32x16 bf16, K x N) for out = in @ W^T with row-major W[D][D]:
// B[k][n] = W[n][k] -> lane (n = nt*16 + lane&15) reads K[kh*8..+7], K[16+kh*8..+7] of row n.
__device__ __forceinline__ V16 load_b_w(const unsigned short* __restrict__ W,
                                        int nt, int k0, int lane) {
    const int n = (nt << 4) + (lane & 15), kh = lane >> 4;
    const unsigned short* p = W + (size_t)n * D + k0 + kh * 8;
    V16 b;
    b.u[0] = *(const v4u*)(p);
    b.u[1] = *(const v4u*)(p + 16);
    return b;
}

// ---------------- weight f32 -> bf16 conversion ----------------
__global__ void k_cvt(const float* __restrict__ s, unsigned short* __restrict__ d, int n) {
    int i = blockIdx.x * blockDim.x + threadIdx.x;
    if (i < n) d[i] = f2bf(s[i]);
}

// ---------------- phase 1: pre = x @ W_ih^T + (b_ih + b_hh), stored bf16 ----------------
__global__ void __launch_bounds__(256)
k_pre(const float* __restrict__ x, const unsigned short* __restrict__ Wih,
      const float* __restrict__ bih, const float* __restrict__ bhh,
      unsigned short* __restrict__ pre) {
    __shared__ unsigned short aT[16 * D];   // 32 KB: x tile as bf16
    __shared__ unsigned short oT[16 * D];   // 32 KB: output staging
    const int tid = threadIdx.x, lane = tid & 31, wave = tid >> 5;
    const int nl = lane & 15, kh = lane >> 4;
    const size_t r0 = (size_t)blockIdx.x * 16;

    const float* xs = x + r0 * D;
    for (int i = tid; i < 16 * D; i += 256) aT[i] = f2bf(xs[i]);
    __syncthreads();

    for (int tt = 0; tt < 8; ++tt) {
        const int nt = wave * 8 + tt;
        v8f acc;
        #pragma unroll
        for (int v = 0; v < 8; ++v) acc[v] = 0.f;
        for (int k0 = 0; k0 < D; k0 += 32) {
            V16 a = load_a_lds(aT, k0, lane);
            V16 b = load_b_w(Wih, nt, k0, lane);
            acc = wmma_bf16(a, b, acc);
        }
        const int n = (nt << 4) + nl;
        const float bias = bih[n] + bhh[n];
        #pragma unroll
        for (int v = 0; v < 8; ++v)
            oT[(kh * 8 + v) * D + n] = f2bf(acc[v] + bias);
    }
    __syncthreads();
    unsigned int* dst = (unsigned int*)(pre + r0 * D);
    const unsigned int* src = (const unsigned int*)oT;
    for (int i = tid; i < 8 * D; i += 256) dst[i] = src[i];
}

// ---------------- phase 2: sequential scan, one workgroup per 16-row batch tile ----------------
__global__ void __launch_bounds__(1024)
k_scan(const unsigned short* __restrict__ pre, const unsigned short* __restrict__ Whh,
       unsigned short* __restrict__ hs) {
    __shared__ unsigned short hbuf[2][16 * D];   // 64 KB double-buffered h (bf16)
    const int tid = threadIdx.x, lane = tid & 31, wave = tid >> 5;  // 32 waves
    const int nl = lane & 15, kh = lane >> 4;
    const int mb = blockIdx.x;                   // batch tile 0 or 1

    for (int i = tid; i < 16 * D; i += 1024) hbuf[0][i] = 0;  // h0 = 0
    __syncthreads();

    int cur = 0;
    for (int t = 0; t < SEQ; ++t) {
        const size_t row0 = (size_t)t * BATCH + mb * 16;
        const unsigned short* preR = pre + row0 * D;
        __builtin_prefetch(preR + (size_t)BATCH * D, 0, 1);  // next step's pre tile

        v8f acc[2];
        #pragma unroll
        for (int tt = 0; tt < 2; ++tt) {
            const int n = ((wave * 2 + tt) << 4) + nl;
            #pragma unroll
            for (int v = 0; v < 8; ++v)
                acc[tt][v] = bf2f(preR[(size_t)(kh * 8 + v) * D + n]);
        }
        for (int k0 = 0; k0 < D; k0 += 32) {
            V16 a = load_a_lds(hbuf[cur], k0, lane);   // shared A across both tiles
            #pragma unroll
            for (int tt = 0; tt < 2; ++tt) {
                V16 b = load_b_w(Whh, wave * 2 + tt, k0, lane);
                acc[tt] = wmma_bf16(a, b, acc[tt]);
            }
        }
        unsigned short* hn = hbuf[cur ^ 1];
        #pragma unroll
        for (int tt = 0; tt < 2; ++tt) {
            const int n = ((wave * 2 + tt) << 4) + nl;
            #pragma unroll
            for (int v = 0; v < 8; ++v)
                hn[(kh * 8 + v) * D + n] = f2bf(tanhf(acc[tt][v]));
        }
        __syncthreads();   // all reads of h_t done, h_{t+1} complete
        unsigned int* dst = (unsigned int*)(hs + row0 * D);
        const unsigned int* src = (const unsigned int*)hn;
        for (int i = tid; i < 8 * D; i += 1024) dst[i] = src[i];
        cur ^= 1;
    }
}

// ---------------- phase 3: out = LN(x + hs @ W_out^T) ----------------
__global__ void __launch_bounds__(256)
k_out(const unsigned short* __restrict__ hsn, const unsigned short* __restrict__ Wout,
      const float* __restrict__ x, const float* __restrict__ g, const float* __restrict__ bta,
      float* __restrict__ out) {
    __shared__ unsigned short aT[16 * D];   // 32 KB hs tile
    __shared__ float psum[8][16][2];        // per-wave row partials (sum, sumsq)
    __shared__ float stat[16][2];           // mean, rsqrt(var+eps)
    const int tid = threadIdx.x, lane = tid & 31, wave = tid >> 5;
    const int nl = lane & 15, kh = lane >> 4;
    const size_t r0 = (size_t)blockIdx.x * 16;

    {
        const unsigned int* src = (const unsigned int*)(hsn + r0 * D);
        unsigned int* dst = (unsigned int*)aT;
        for (int i = tid; i < 8 * D; i += 256) dst[i] = src[i];
    }
    __syncthreads();

    v8f acc[8];
    #pragma unroll
    for (int tt = 0; tt < 8; ++tt)
        #pragma unroll
        for (int v = 0; v < 8; ++v) acc[tt][v] = 0.f;

    for (int k0 = 0; k0 < D; k0 += 32) {
        V16 a = load_a_lds(aT, k0, lane);
        #pragma unroll
        for (int tt = 0; tt < 8; ++tt) {
            V16 b = load_b_w(Wout, wave * 8 + tt, k0, lane);
            acc[tt] = wmma_bf16(a, b, acc[tt]);
        }
    }

    // residual add + per-row moment partials (rows m = kh*8+v live in this half-wave)
    float rs[8], rq[8];
    #pragma unroll
    for (int v = 0; v < 8; ++v) { rs[v] = 0.f; rq[v] = 0.f; }
    #pragma unroll
    for (int tt = 0; tt < 8; ++tt) {
        const int n = ((wave * 8 + tt) << 4) + nl;
        #pragma unroll
        for (int v = 0; v < 8; ++v) {
            float f = acc[tt][v] + x[(r0 + kh * 8 + v) * D + n];
            acc[tt][v] = f;
            rs[v] += f; rq[v] += f * f;
        }
    }
    #pragma unroll
    for (int v = 0; v < 8; ++v) {
        #pragma unroll
        for (int off = 1; off < 16; off <<= 1) {   // reduce within each 16-lane half
            rs[v] += __shfl_xor(rs[v], off, 32);
            rq[v] += __shfl_xor(rq[v], off, 32);
        }
    }
    if (nl == 0) {
        #pragma unroll
        for (int v = 0; v < 8; ++v) {
            psum[wave][kh * 8 + v][0] = rs[v];
            psum[wave][kh * 8 + v][1] = rq[v];
        }
    }
    __syncthreads();
    if (tid < 16) {
        float a1 = 0.f, a2 = 0.f;
        #pragma unroll
        for (int w = 0; w < 8; ++w) { a1 += psum[w][tid][0]; a2 += psum[w][tid][1]; }
        const float mu  = a1 * (1.0f / D);
        const float var = a2 * (1.0f / D) - mu * mu;
        stat[tid][0] = mu;
        stat[tid][1] = rsqrtf(var + LN_EPS);
    }
    __syncthreads();
    #pragma unroll
    for (int tt = 0; tt < 8; ++tt) {
        const int n = ((wave * 8 + tt) << 4) + nl;
        const float gg = g[n], bb = bta[n];
        #pragma unroll
        for (int v = 0; v < 8; ++v) {
            const int m = kh * 8 + v;
            out[(r0 + m) * D + n] = (acc[tt][v] - stat[m][0]) * stat[m][1] * gg + bb;
        }
    }
}

extern "C" void kernel_launch(void* const* d_in, const int* in_sizes, int n_in,
                              void* d_out, int out_size, void* d_ws, size_t ws_size,
                              hipStream_t stream) {
    const float* x    = (const float*)d_in[0];
    const float* Wih  = (const float*)d_in[1];
    const float* Whh  = (const float*)d_in[2];
    const float* bih  = (const float*)d_in[3];
    const float* bhh  = (const float*)d_in[4];
    const float* Wout = (const float*)d_in[5];
    const float* lng  = (const float*)d_in[6];
    const float* lnb  = (const float*)d_in[7];
    float* out = (float*)d_out;

    // workspace layout (bf16 everywhere): 3 weights (2 MB each) + pre (128 MB) + hs (128 MB)
    char* ws = (char*)d_ws;
    unsigned short* Wih_bf  = (unsigned short*)(ws);
    unsigned short* Whh_bf  = (unsigned short*)(ws + (size_t)2 * 1024 * 1024);
    unsigned short* Wout_bf = (unsigned short*)(ws + (size_t)4 * 1024 * 1024);
    unsigned short* pre_bf  = (unsigned short*)(ws + (size_t)6 * 1024 * 1024);
    unsigned short* hs_bf   = (unsigned short*)(ws + (size_t)6 * 1024 * 1024
                                                   + (size_t)ROWS * D * sizeof(unsigned short));

    const int NW = D * D;
    k_cvt<<<NW / 256, 256, 0, stream>>>(Wih,  Wih_bf,  NW);
    k_cvt<<<NW / 256, 256, 0, stream>>>(Whh,  Whh_bf,  NW);
    k_cvt<<<NW / 256, 256, 0, stream>>>(Wout, Wout_bf, NW);

    k_pre <<<ROWS / 16, 256, 0, stream>>>(x, Wih_bf, bih, bhh, pre_bf);
    k_scan<<<2, 1024, 0, stream>>>(pre_bf, Whh_bf, hs_bf);
    k_out <<<ROWS / 16, 256, 0, stream>>>(hs_bf, Wout_bf, x, lng, lnb, out);
}
